// MultiheadAttention_14216341750216
// MI455X (gfx1250) — matrix operand order repaired and checked by reading, not compiled, not run on hardware
//
#include <hip/hip_runtime.h>
#include <hip/hip_bf16.h>

// MI455X / gfx1250, wave32, WMMA f32_16x16x32_f16 everywhere.
// Attention staging uses GLOBAL_LOAD_ASYNC_TO_LDS (ASYNCcnt) + double buffering.

typedef __attribute__((ext_vector_type(16))) _Float16 v16h;
typedef __attribute__((ext_vector_type(8)))  _Float16 v8h;
typedef __attribute__((ext_vector_type(4)))  _Float16 v4h;
typedef __attribute__((ext_vector_type(8)))  float    v8f;
typedef __attribute__((ext_vector_type(4)))  int      v4i;

#define L_SEQ 4096
#define E_DIM 768
#define H_HEADS 12
#define D_HEAD 64
#define MW (L_SEQ / 32)   // mask bit-words per row = 128
static constexpr float INV_SCALE = 0.03608439182435161f; // 1/sqrt(768)

#if defined(__has_builtin)
#  if __has_builtin(__builtin_amdgcn_global_load_async_to_lds_b128)
#    define ASYNC_LDS 1
#  endif
#endif
#ifndef ASYNC_LDS
#  define ASYNC_LDS 0
#endif

typedef __attribute__((address_space(1))) v4i gv4i;   // global int4*
typedef __attribute__((address_space(3))) v4i lv4i;   // LDS int4*

__device__ __forceinline__ void async_cp16(const _Float16* g, _Float16* l) {
#if ASYNC_LDS
  __builtin_amdgcn_global_load_async_to_lds_b128((gv4i*)(_Float16*)g, (lv4i*)l, 0, 0);
#else
  *(v8h*)l = *(const v8h*)g;
#endif
}

__device__ __forceinline__ void wait_async0() {
#if ASYNC_LDS
#  if __has_builtin(__builtin_amdgcn_s_wait_asynccnt)
  __builtin_amdgcn_s_wait_asynccnt(0);
#  else
  asm volatile("s_wait_asynccnt 0" ::: "memory");
#  endif
#endif
}

union ABu { v16h v; v8h h[2]; };

// A fragment (16x32 f16, MxK): lane row = lane&15.
// element i -> K = (i<8 ? i : i+8) + 8*hi.  p = &src[row*rs + k0 + 8*hi]
__device__ __forceinline__ v16h load_a16(const _Float16* p) {
  ABu u;
  u.h[0] = *(const v8h*)(p);
  u.h[1] = *(const v8h*)(p + 16);
  return u.v;
}
// B fragment (32x16 f16, KxN): lane col = lane&15.
// element i -> K = (i<8 ? i : i+8) + 8*hi.  p = &src[col*rs + k0 + 8*hi]  (src stored N x K)
__device__ __forceinline__ v16h load_b16(const _Float16* p) {
  ABu u;
  u.h[0] = *(const v8h*)(p);
  u.h[1] = *(const v8h*)(p + 16);
  return u.v;
}

__device__ __forceinline__ v8f wmma32(v16h a, v16h b, v8f c) {
  return __builtin_amdgcn_wmma_f32_16x16x32_f16(false, a, false, b, (short)0, c,
                                                false, false);
}

// ---------------- stage 1: fp32 -> f16 convert ----------------
__global__ void __launch_bounds__(256) cvt_f16(const float* __restrict__ in,
                                               _Float16* __restrict__ out, int n) {
  int i = blockIdx.x * 256 + threadIdx.x;
  if (i < n / 4) {
    float4 f = ((const float4*)in)[i];
    v4h h;
    h[0] = (_Float16)f.x; h[1] = (_Float16)f.y;
    h[2] = (_Float16)f.z; h[3] = (_Float16)f.w;
    ((v4h*)out)[i] = h;
  }
}

// ---------------- stage 2: weight transpose+convert: Wt[e][k] = W[k][e] ----
__global__ void __launch_bounds__(256) wtx(const float* __restrict__ W,
                                           _Float16* __restrict__ Wt) {
  int idx = blockIdx.x * 256 + threadIdx.x;
  if (idx >= E_DIM * E_DIM) return;
  int e = idx / E_DIM, k = idx % E_DIM;
  Wt[idx] = (_Float16)W[k * E_DIM + e];   // coalesced writes
}

// ---------------- stage 2b: pack mask into bits (1 bit per (q,s)) ----------
__global__ void __launch_bounds__(256) packmask(const int* __restrict__ mask,
                                                unsigned* __restrict__ bits) {
  int i = blockIdx.x * 256 + threadIdx.x;           // 4096 * 128 words
  if (i >= L_SEQ * MW) return;
  int q = i >> 7, w = i & (MW - 1);
  const int4* p4 = (const int4*)(mask + (size_t)q * L_SEQ + w * 32);
  unsigned b = 0;
#pragma unroll
  for (int j = 0; j < 8; ++j) {
    int4 v = p4[j];
    b |= (v.x != 0 ? 1u : 0u) << (j * 4 + 0);
    b |= (v.y != 0 ? 1u : 0u) << (j * 4 + 1);
    b |= (v.z != 0 ? 1u : 0u) << (j * 4 + 2);
    b |= (v.w != 0 ? 1u : 0u) << (j * 4 + 3);
  }
  bits[i] = b;
}

// ---------------- stage 3/5: WMMA GEMM, C(4096x768) = A(4096x768) * Bt^T ----
// One wave computes a 32x64 tile (2 row-tiles reuse each B fragment).
// MODE 0: f16 out as [H][L][64]. MODE 1: f16 out as [H][64][L]. MODE 2: f32+bias.
template <int MODE>
__global__ void __launch_bounds__(128) gemm768(const _Float16* __restrict__ A,
                                               const _Float16* __restrict__ Bt,
                                               _Float16* __restrict__ outH,
                                               float* __restrict__ outF,
                                               const float* __restrict__ bias) {
  const int lane = threadIdx.x & 31;
  const int wid  = threadIdx.x >> 5;
  const int wg   = blockIdx.x * 4 + wid;   // 1536 waves = 128 row-tiles x 12 col-groups
  const int rowT = wg & 127;
  const int cg   = wg >> 7;
  const int hi = lane >> 4, ln = lane & 15;
  const int row0 = rowT * 32 + ln;

  v8f acc[2][4] = {};
  for (int k0 = 0; k0 < E_DIM; k0 += 32) {
    v16h a0 = load_a16(A + (row0)      * E_DIM + k0 + hi * 8);
    v16h a1 = load_a16(A + (row0 + 16) * E_DIM + k0 + hi * 8);
#pragma unroll
    for (int dt = 0; dt < 4; ++dt) {
      int col = cg * 64 + dt * 16 + ln;
      v16h b = load_b16(Bt + col * E_DIM + k0 + hi * 8);
      acc[0][dt] = wmma32(a0, b, acc[0][dt]);
      acc[1][dt] = wmma32(a1, b, acc[1][dt]);
    }
  }
#pragma unroll
  for (int mt = 0; mt < 2; ++mt) {
#pragma unroll
    for (int dt = 0; dt < 4; ++dt) {
      int col = cg * 64 + dt * 16 + ln;
#pragma unroll
      for (int r = 0; r < 8; ++r) {
        int m = rowT * 32 + mt * 16 + r + hi * 8;   // C/D layout: M = r + 8*hi
        float v = acc[mt][dt][r];
        if (MODE == 0) {
          int h = col >> 6, d = col & 63;
          outH[(h * L_SEQ + m) * D_HEAD + d] = (_Float16)v;
        } else if (MODE == 1) {
          int h = col >> 6, d = col & 63;
          outH[(h * D_HEAD + d) * L_SEQ + m] = (_Float16)v;
        } else {
          outF[(size_t)m * E_DIM + col] = v + bias[col];
        }
      }
    }
  }
}

// ---------------- stage 4: flash attention ----------------
// grid = H * (L/128) blocks of 256 threads (8 waves, one 16-query tile each);
// block double-buffers 64-key K / V^T tiles in LDS via async-to-LDS loads.
__global__ void __launch_bounds__(256) attn_kernel(const _Float16* __restrict__ Qh,
                                                   const _Float16* __restrict__ Kh,
                                                   const _Float16* __restrict__ Vt,
                                                   const unsigned* __restrict__ mbits,
                                                   _Float16* __restrict__ ctx) {
  __shared__ __align__(16) _Float16 ksh[2][64 * 64];     // [key][d]   2x8KB
  __shared__ __align__(16) _Float16 vsh[2][64 * 64];     // [d][key]   2x8KB
  __shared__ __align__(16) _Float16 psh[8][2][16 * 32];  // per-wave P 16KB

  const int t = threadIdx.x;
  const int lane = t & 31, wid = t >> 5;
  const int head = blockIdx.x >> 5;            // 12 heads
  const int qb   = blockIdx.x & 31;            // 32 blocks of 128 queries
  const int q0 = qb * 128 + wid * 16;
  const int hi = lane >> 4, ln = lane & 15;

  const _Float16* qp = Qh + (size_t)head * L_SEQ * D_HEAD;
  const _Float16* kp = Kh + (size_t)head * L_SEQ * D_HEAD;
  const _Float16* vp = Vt + (size_t)head * D_HEAD * L_SEQ;

  // coop-staging indices: 256 threads move 8KB+8KB per tile
  const int sr = t >> 2, ss = (t & 3) * 16;
  auto stage = [&](int buf, int s0) {
    async_cp16(kp + (s0 + sr) * D_HEAD + ss,     &ksh[buf][sr * 64 + ss]);
    async_cp16(kp + (s0 + sr) * D_HEAD + ss + 8, &ksh[buf][sr * 64 + ss + 8]);
    async_cp16(vp + sr * L_SEQ + s0 + ss,        &vsh[buf][sr * 64 + ss]);
    async_cp16(vp + sr * L_SEQ + s0 + ss + 8,    &vsh[buf][sr * 64 + ss + 8]);
  };

  // Q A-fragments for D=64 (two K=32 slabs), kept in registers for whole loop
  v16h qa0 = load_a16(qp + (q0 + ln) * D_HEAD + 0  + hi * 8);
  v16h qa1 = load_a16(qp + (q0 + ln) * D_HEAD + 32 + hi * 8);

  v8f o[4] = {};
  float mrow[8], ssum[8];        // ssum is a per-lane PARTIAL sum (reduced at end)
#pragma unroll
  for (int r = 0; r < 8; ++r) { mrow[r] = -3.0e38f; ssum[r] = 0.f; }

  stage(0, 0);                                  // prologue: fill buffer 0

  for (int it = 0; it < L_SEQ / 64; ++it) {
    const int s0 = it * 64;
    const int cur = it & 1;
    wait_async0();       // this wave's stage of `cur` complete
    __syncthreads();     // all waves staged `cur`; prior reads of `cur^1` retired
    if (it + 1 < L_SEQ / 64) stage(cur ^ 1, s0 + 64);  // overlap with compute

    // scores: four 16x16 tiles over 64 keys, K-dim = D = 64 (2 WMMAs each)
    v8f c[4];
#pragma unroll
    for (int nt = 0; nt < 4; ++nt) {
      v16h b0 = load_b16(&ksh[cur][(nt * 16 + ln) * 64 + 0  + hi * 8]);
      v16h b1 = load_b16(&ksh[cur][(nt * 16 + ln) * 64 + 32 + hi * 8]);
      v8f cc = {};
      cc = wmma32(qa0, b0, cc);
      cc = wmma32(qa1, b1, cc);
      c[nt] = cc;
    }

    // mask + scale + online softmax (row = r + 8*hi, col tiles ln + 16*nt)
#pragma unroll
    for (int r = 0; r < 8; ++r) {
      int qrow = q0 + r + hi * 8;
      uint2 mp = *(const uint2*)(mbits + (size_t)qrow * MW + (s0 >> 5));
      float x[4];
      x[0] = (((mp.x >> ln)        & 1u) ? (float)c[0][r] : -1e20f) * INV_SCALE;
      x[1] = (((mp.x >> (16 + ln)) & 1u) ? (float)c[1][r] : -1e20f) * INV_SCALE;
      x[2] = (((mp.y >> ln)        & 1u) ? (float)c[2][r] : -1e20f) * INV_SCALE;
      x[3] = (((mp.y >> (16 + ln)) & 1u) ? (float)c[3][r] : -1e20f) * INV_SCALE;
      float smax = fmaxf(fmaxf(x[0], x[1]), fmaxf(x[2], x[3]));
      smax = fmaxf(smax, __shfl_xor(smax, 1));
      smax = fmaxf(smax, __shfl_xor(smax, 2));
      smax = fmaxf(smax, __shfl_xor(smax, 4));
      smax = fmaxf(smax, __shfl_xor(smax, 8));
      float mnew = fmaxf(mrow[r], smax);
      float corr = __expf(mrow[r] - mnew);
      float p0 = __expf(x[0] - mnew);
      float p1 = __expf(x[1] - mnew);
      float p2 = __expf(x[2] - mnew);
      float p3 = __expf(x[3] - mnew);
      ssum[r] = ssum[r] * corr + ((p0 + p1) + (p2 + p3));  // lane-local partial
      mrow[r] = mnew;
      o[0][r] *= corr; o[1][r] *= corr; o[2][r] *= corr; o[3][r] *= corr;
      int prow = r + hi * 8;
      psh[wid][0][prow * 32 + ln]      = (_Float16)p0;  // D-layout -> A-tile LDS
      psh[wid][0][prow * 32 + 16 + ln] = (_Float16)p1;
      psh[wid][1][prow * 32 + ln]      = (_Float16)p2;
      psh[wid][1][prow * 32 + 16 + ln] = (_Float16)p3;
    }
    // psh[wid] is wave-private; same-wave DS ops execute in order (ISA 7.3),
    // so only a compiler reordering fence is needed (no workgroup barrier).
    asm volatile("" ::: "memory");

    // PV: A = P (two 16x32 tiles), B = V^T tiles; 8 WMMAs cover D=64 x 64 keys
#pragma unroll
    for (int kf = 0; kf < 2; ++kf) {
      v16h pa = load_a16(&psh[wid][kf][ln * 32 + hi * 8]);
#pragma unroll
      for (int dt = 0; dt < 4; ++dt) {
        v16h vb = load_b16(&vsh[cur][(dt * 16 + ln) * 64 + kf * 32 + hi * 8]);
        o[dt] = wmma32(pa, vb, o[dt]);
      }
    }
  }

  // epilogue: reduce per-lane partial sums across the 16-lane group, normalize
#pragma unroll
  for (int r = 0; r < 8; ++r) {
    float s = ssum[r];
    s += __shfl_xor(s, 1);
    s += __shfl_xor(s, 2);
    s += __shfl_xor(s, 4);
    s += __shfl_xor(s, 8);
    float inv = 1.0f / s;
    int qrow = q0 + r + hi * 8;
#pragma unroll
    for (int dt = 0; dt < 4; ++dt) {
      ctx[(size_t)qrow * E_DIM + head * 64 + dt * 16 + ln] =
          (_Float16)(o[dt][r] * inv);
    }
  }
}

// ---------------- driver ----------------
extern "C" void kernel_launch(void* const* d_in, const int* in_sizes, int n_in,
                              void* d_out, int out_size, void* d_ws, size_t ws_size,
                              hipStream_t stream) {
  const float* query = (const float*)d_in[0];
  const float* key   = (const float*)d_in[1];
  const float* value = (const float*)d_in[2];
  const int*   mask  = (const int*)d_in[3];
  const float* Wq = (const float*)d_in[4];
  const float* Wk = (const float*)d_in[5];
  const float* Wv = (const float*)d_in[6];
  const float* Wo = (const float*)d_in[7];
  const float* bo = (const float*)d_in[8];

  const size_t LE = (size_t)L_SEQ * E_DIM;   // 3,145,728 halves
  const size_t EE = (size_t)E_DIM * E_DIM;   //   589,824 halves
  _Float16* x16q = (_Float16*)d_ws;
  _Float16* x16k = x16q + LE;
  _Float16* x16v = x16k + LE;
  _Float16* wqt  = x16v + LE;
  _Float16* wkt  = wqt + EE;
  _Float16* wvt  = wkt + EE;
  _Float16* wot  = wvt + EE;
  _Float16* qh   = wot + EE;   // [H][L][64]
  _Float16* kh   = qh + LE;    // [H][S][64]
  _Float16* vt   = kh + LE;    // [H][64][S]
  _Float16* ctx  = vt + LE;    // [L][E]
  unsigned* mbits = (unsigned*)(ctx + LE);   // 4096 x 128 u32 = 2MB

  int nb = (int)(LE / 4 + 255) / 256;
  cvt_f16<<<nb, 256, 0, stream>>>(query, x16q, (int)LE);
  cvt_f16<<<nb, 256, 0, stream>>>(key,   x16k, (int)LE);
  cvt_f16<<<nb, 256, 0, stream>>>(value, x16v, (int)LE);

  int wb = (int)((EE + 255) / 256);
  wtx<<<wb, 256, 0, stream>>>(Wq, wqt);
  wtx<<<wb, 256, 0, stream>>>(Wk, wkt);
  wtx<<<wb, 256, 0, stream>>>(Wv, wvt);
  wtx<<<wb, 256, 0, stream>>>(Wo, wot);

  int mbn = L_SEQ * MW;                       // 524288 words
  packmask<<<(mbn + 255) / 256, 256, 0, stream>>>(mask, mbits);

  gemm768<0><<<384, 128, 0, stream>>>(x16q, wqt, qh, nullptr, nullptr);
  gemm768<0><<<384, 128, 0, stream>>>(x16k, wkt, kh, nullptr, nullptr);
  gemm768<1><<<384, 128, 0, stream>>>(x16v, wvt, vt, nullptr, nullptr);

  attn_kernel<<<H_HEADS * 32, 256, 0, stream>>>(qh, kh, vt, mbits, ctx);

  gemm768<2><<<384, 128, 0, stream>>>(ctx, wot, nullptr, (float*)d_out, bo);
}
